// DspGNN_75239237091507
// MI455X (gfx1250) — matrix operand order repaired and checked by reading, not compiled, not run on hardware
//
#include <hip/hip_runtime.h>
#include <hip/hip_bf16.h>
#include <stdint.h>

// ---------------- CDNA5 WMMA types ----------------
typedef __bf16  bf16x16 __attribute__((ext_vector_type(16)));
typedef float   floatx8 __attribute__((ext_vector_type(8)));

union FragAB {
    bf16x16        v;
    unsigned int   d[8];
    unsigned short u[16];
};

static __device__ __forceinline__ unsigned short f2bf(float f) {
    unsigned int u = __float_as_uint(f);
    u += 0x7FFFu + ((u >> 16) & 1u);   // round-to-nearest-even
    return (unsigned short)(u >> 16);
}
static __device__ __forceinline__ float bf2f(unsigned short b) {
    return __uint_as_float(((unsigned int)b) << 16);
}

// A fragment: 16x32 bf16, row-major A[lda], row = tile_row*16 + lane%16.
// ISA 7.12.2: lanes 0-15 hold K = {0..7,16..23}, lanes 16-31 hold K = {8..15,24..31}.
// The 8 dwords are two 16B-contiguous runs -> compiler emits 2x global_load_b128.
static __device__ __forceinline__ bf16x16
load_a_frag(const unsigned short* __restrict__ A, int lda, int row, int grp, int k0) {
    FragAB f;
    const unsigned short* p = A + (size_t)row * lda + k0 + grp * 8;
#pragma unroll
    for (int r = 0; r < 8; ++r) {
        int k = (r < 4) ? (2 * r) : (16 + 2 * (r - 4));
        f.d[r] = *(const unsigned int*)(p + k);
    }
    return f.v;
}

// B fragment from pre-packed layout: tile (kt,nt) stored as 32 lanes x 8 dwords,
// contiguous per lane -> 2x global_load_b128 per fragment.
static __device__ __forceinline__ bf16x16
load_b_packed(const unsigned int* __restrict__ Bpk, int NT, int kt, int nt, int lane) {
    FragAB f;
    const unsigned int* p = Bpk + (((size_t)kt * NT + nt) * 32 + lane) * 8;
#pragma unroll
    for (int r = 0; r < 8; ++r) f.d[r] = p[r];
    return f.v;
}

static __device__ __forceinline__ floatx8
wmma_bf16(bf16x16 a, bf16x16 b, floatx8 c) {
    return __builtin_amdgcn_wmma_f32_16x16x32_bf16(false, a, false, b, (short)0, c, false, false);
}

// ---------------- weight pre-pack: f32 [Kdim x Ncols] (or transposed src) -> B fragments ----
// One wave per 32x16 K-tile. Lane layout: col = nt*16 + lane%16; grp = lane/16;
// dword r holds bf16 pair (K = kt*32 + grp*8 + (r<4 ? 2r : 16 + 2(r-4)), K+1).
__global__ void k_pack_b(const float* __restrict__ src, int Kdim, int Ncols, int transp,
                         unsigned int* __restrict__ dst) {
    int KT = Kdim >> 5, NT = Ncols >> 4;
    int tile = blockIdx.x * 4 + (threadIdx.x >> 5);
    if (tile >= KT * NT) return;
    int lane = threadIdx.x & 31;
    int kt = tile / NT, nt = tile % NT;
    int col = nt * 16 + (lane & 15);
    int grp = lane >> 4;
    unsigned int* out = dst + ((size_t)tile * 32 + lane) * 8;
#pragma unroll
    for (int r = 0; r < 8; ++r) {
        int K = kt * 32 + grp * 8 + ((r < 4) ? (2 * r) : (16 + 2 * (r - 4)));
        float f0, f1;
        if (transp) {            // logical B[k][n] = src[n*Kdim + k]
            f0 = src[(size_t)col * Kdim + K];
            f1 = src[(size_t)col * Kdim + K + 1];
        } else {                 // logical B[k][n] = src[k*Ncols + n]
            f0 = src[(size_t)K * Ncols + col];
            f1 = src[(size_t)(K + 1) * Ncols + col];
        }
        out[r] = (unsigned int)f2bf(f0) | ((unsigned int)f2bf(f1) << 16);
    }
}

// ---------------- utility kernels ----------------
__global__ void k_zero_f32(float* p, size_t n) {
    size_t i = (size_t)blockIdx.x * blockDim.x + threadIdx.x;
    if (i < n) p[i] = 0.0f;
}
__global__ void k_zero_u16(unsigned short* p, size_t n) {
    size_t i = (size_t)blockIdx.x * blockDim.x + threadIdx.x;
    if (i < n) p[i] = 0;
}

// Hk f32 [N,128] -> bf16 copies: Hk_bf [N,128], cat cols 0..127 (ld 896), catpl cols 0..127 (ld 512)
__global__ void k_prep_Hk(const float* __restrict__ Hk, unsigned short* __restrict__ Hk_bf,
                          unsigned short* __restrict__ cat, unsigned short* __restrict__ catpl,
                          size_t total) {
    size_t i = (size_t)blockIdx.x * blockDim.x + threadIdx.x;
    if (i >= total) return;
    size_t n = i >> 7; int d = (int)(i & 127);
    unsigned short b = f2bf(Hk[i]);
    Hk_bf[i] = b;
    cat[n * 896 + d] = b;
    catpl[n * 512 + d] = b;
}

// ---------------- WMMA GEMM: C = act( A*B [+ A2*B2] + bias ), bf16 out ----------------
// block = 128 threads (4 waves). Each wave computes TWO adjacent 16x16 col tiles
// (A fragment reused across both). grid.x = M/16; grid.y covers Ncols in chunks of 128.
__global__ void k_gemm_bf16(const unsigned short* __restrict__ A, int lda,
                            const unsigned int* __restrict__ Bpk,
                            const unsigned short* __restrict__ A2,
                            const unsigned int* __restrict__ B2pk,
                            const float* __restrict__ bias, int relu_act,
                            unsigned short* __restrict__ outBf, int ldc, int colOff,
                            int Ncols, int Kdim) {
    int wave = threadIdx.x >> 5;
    int lane = threadIdx.x & 31;
    int grp  = lane >> 4;
    int NT   = Ncols >> 4;
    int tileM = blockIdx.x;
    int t0 = blockIdx.y * 8 + wave * 2;     // first of two col tiles (Ncols % 128 == 0)
    int row = tileM * 16 + (lane & 15);

    floatx8 acc0 = {0.f, 0.f, 0.f, 0.f, 0.f, 0.f, 0.f, 0.f};
    floatx8 acc1 = {0.f, 0.f, 0.f, 0.f, 0.f, 0.f, 0.f, 0.f};
    for (int k0 = 0; k0 < Kdim; k0 += 32) {
        if (k0 + 32 < Kdim)
            __builtin_prefetch(A + (size_t)row * lda + k0 + 32, 0, 1);   // global_prefetch_b8
        bf16x16 a = load_a_frag(A, lda, row, grp, k0);
        acc0 = wmma_bf16(a, load_b_packed(Bpk, NT, k0 >> 5, t0,     lane), acc0);
        acc1 = wmma_bf16(a, load_b_packed(Bpk, NT, k0 >> 5, t0 + 1, lane), acc1);
    }
    if (A2) {                                // wave-uniform branch: EXEC stays all-1s
        for (int k0 = 0; k0 < Kdim; k0 += 32) {
            bf16x16 a = load_a_frag(A2, lda, row, grp, k0);
            acc0 = wmma_bf16(a, load_b_packed(B2pk, NT, k0 >> 5, t0,     lane), acc0);
            acc1 = wmma_bf16(a, load_b_packed(B2pk, NT, k0 >> 5, t0 + 1, lane), acc1);
        }
    }
#pragma unroll
    for (int t = 0; t < 2; ++t) {
        int col = (t0 + t) * 16 + (lane & 15);
        float bv = bias ? bias[col] : 0.0f;
        floatx8 acc = t ? acc1 : acc0;
#pragma unroll
        for (int r = 0; r < 8; ++r) {
            float v = acc[r] + bv;
            if (relu_act) v = fmaxf(v, 0.0f);
            int rr = tileM * 16 + r + grp * 8;
            outBf[(size_t)rr * ldc + colOff + col] = f2bf(v);
        }
    }
}

// ---------------- edge scatter: agg[dst] += ew * h[src]; cnt[dst] += 1 ----------------
// one wave (32 lanes) per edge, 4 dims/lane; block = 256 = 8 waves
__global__ void k_edge_scatter(const int* __restrict__ src, const int* __restrict__ dst,
                               const float* __restrict__ ew, const unsigned short* __restrict__ h,
                               float* __restrict__ agg, float* __restrict__ cnt, int E) {
    int e = blockIdx.x * 8 + (threadIdx.x >> 5);
    if (e >= E) return;
    int lane = threadIdx.x & 31;
    int s = src[e], d = dst[e];
    float w = ew[e];
    const unsigned int* hp = (const unsigned int*)(h + (size_t)s * 128);  // 2 bf16 per dword
    float* ap = agg + (size_t)d * 128;
#pragma unroll
    for (int i = 0; i < 2; ++i) {
        unsigned int pr = hp[lane + i * 32];
        int c = 2 * (lane + i * 32);
        atomicAdd(ap + c,     w * bf2f((unsigned short)(pr & 0xFFFFu)));
        atomicAdd(ap + c + 1, w * bf2f((unsigned short)(pr >> 16)));
    }
    if (lane == 0) atomicAdd(cnt + d, 1.0f);
}

// agg_bf = bf16( agg / max(cnt,1) ) ; one block (128 thr) per node
__global__ void k_normalize(const float* __restrict__ agg, const float* __restrict__ cnt,
                            unsigned short* __restrict__ agg_bf) {
    size_t n = blockIdx.x;
    float inv = 1.0f / fmaxf(cnt[n], 1.0f);
    agg_bf[n * 128 + threadIdx.x] = f2bf(agg[n * 128 + threadIdx.x] * inv);
}

// ---------------- fused LSTM step ----------------
// block = 128 thr (4 waves) per 16-node tile. gates[16x512] via WMMA into LDS, then cell.
// All 4 K-tiles of the x/h A fragments are hoisted to registers and reused across the
// 8 gate column tiles each wave owns (processed in pairs).
__global__ void k_lstm_step(const unsigned short* __restrict__ x,      // [N,128] bf16
                            const unsigned int* __restrict__ WihPk,    // packed [128x512]
                            const unsigned int* __restrict__ WhhPk,    // packed [128x512]
                            const float* __restrict__ bih, const float* __restrict__ bhh,
                            unsigned short* __restrict__ h,            // [N,128] bf16 (in/out)
                            float* __restrict__ c,                     // [N,128] f32 (in/out)
                            float* __restrict__ out, int last) {
    __shared__ float gates[16 * 512];
    int wave = threadIdx.x >> 5;
    int lane = threadIdx.x & 31;
    int grp  = lane >> 4;
    int tileM = blockIdx.x;
    int row = tileM * 16 + (lane & 15);

    bf16x16 axs[4], ahs[4];
#pragma unroll
    for (int kk = 0; kk < 4; ++kk) {
        axs[kk] = load_a_frag(x, 128, row, grp, kk * 32);
        ahs[kk] = load_a_frag(h, 128, row, grp, kk * 32);
    }

#pragma unroll
    for (int tp = 0; tp < 4; ++tp) {
        int t0 = wave * 8 + tp * 2;          // two of this wave's 8 col tiles (NT = 32)
        floatx8 acc0 = {0.f, 0.f, 0.f, 0.f, 0.f, 0.f, 0.f, 0.f};
        floatx8 acc1 = {0.f, 0.f, 0.f, 0.f, 0.f, 0.f, 0.f, 0.f};
#pragma unroll
        for (int kk = 0; kk < 4; ++kk) {
            acc0 = wmma_bf16(axs[kk], load_b_packed(WihPk, 32, kk, t0,     lane), acc0);
            acc1 = wmma_bf16(axs[kk], load_b_packed(WihPk, 32, kk, t0 + 1, lane), acc1);
            acc0 = wmma_bf16(ahs[kk], load_b_packed(WhhPk, 32, kk, t0,     lane), acc0);
            acc1 = wmma_bf16(ahs[kk], load_b_packed(WhhPk, 32, kk, t0 + 1, lane), acc1);
        }
#pragma unroll
        for (int t = 0; t < 2; ++t) {
            int col = (t0 + t) * 16 + (lane & 15);
            float bv = bih[col] + bhh[col];
            floatx8 acc = t ? acc1 : acc0;
#pragma unroll
            for (int r = 0; r < 8; ++r)
                gates[(r + grp * 8) * 512 + col] = acc[r] + bv;
        }
    }
    __syncthreads();

    // cell: 16 nodes x 128 dims = 2048 values / 128 threads
    for (int idx = threadIdx.x; idx < 2048; idx += 128) {
        int rloc = idx >> 7, d = idx & 127;
        size_t node = (size_t)tileM * 16 + rloc;
        float gi = gates[rloc * 512 + d];
        float gf = gates[rloc * 512 + 128 + d];
        float gg = gates[rloc * 512 + 256 + d];
        float go = gates[rloc * 512 + 384 + d];
        float si = 1.0f / (1.0f + __expf(-gi));
        float sf = 1.0f / (1.0f + __expf(-gf));
        float so = 1.0f / (1.0f + __expf(-go));
        float cn = sf * c[node * 128 + d] + si * tanhf(gg);
        float hn = so * tanhf(cn);
        c[node * 128 + d] = cn;
        h[node * 128 + d] = f2bf(hn);
        if (last) out[node * 128 + d] = fmaxf(hn, 0.0f);
    }
}

// ---------------- host orchestration ----------------
extern "C" void kernel_launch(void* const* d_in, const int* in_sizes, int n_in,
                              void* d_out, int out_size, void* d_ws, size_t ws_size,
                              hipStream_t stream) {
    const int K = 3, N = 30000, D = 128, NCONV = 6, E = 480000;
    const int MT = N / 16;                     // 1875 row tiles (exact)

    const float* H     = (const float*)d_in[0];
    const int*   el    = (const int*)  d_in[1];
    const float* spec  = (const float*)d_in[2];
    const float* Wp    = (const float*)d_in[3];
    const float* bp    = (const float*)d_in[4];
    const float* Wrel  = (const float*)d_in[5];
    const float* Wroot = (const float*)d_in[6];
    const float* Wcomb = (const float*)d_in[7];
    const float* bcomb = (const float*)d_in[8];
    const float* Wpl   = (const float*)d_in[9];
    const float* bpl   = (const float*)d_in[10];
    const float* Wih   = (const float*)d_in[11];
    const float* Whh   = (const float*)d_in[12];
    const float* bih   = (const float*)d_in[13];
    const float* bhh   = (const float*)d_in[14];
    float* out = (float*)d_out;

    // workspace carve (256B aligned)
    size_t off = 0;
    auto alloc = [&](size_t bytes) -> void* {
        void* p = (char*)d_ws + off;
        off += (bytes + 255) & ~(size_t)255;
        return p;
    };
    // packed weights: same footprint as bf16 (2 B/elem)
    unsigned int* Wp_pk    = (unsigned int*)alloc((size_t)NCONV * D * D * 2);   // 8192 dwords each
    unsigned int* Wrel_pk  = (unsigned int*)alloc((size_t)NCONV * D * D * 2);
    unsigned int* Wroot_pk = (unsigned int*)alloc((size_t)NCONV * D * D * 2);
    unsigned int* Wcomb_pk = (unsigned int*)alloc((size_t)896 * 384 * 2);
    unsigned int* Wpl_pk   = (unsigned int*)alloc((size_t)512 * 128 * 2);
    unsigned int* Wih_pk   = (unsigned int*)alloc((size_t)128 * 512 * 2);
    unsigned int* Whh_pk   = (unsigned int*)alloc((size_t)128 * 512 * 2);
    unsigned short* Hk_bf    = (unsigned short*)alloc((size_t)N * D * 2);
    unsigned short* hproj_bf = (unsigned short*)alloc((size_t)N * D * 2);
    unsigned short* agg_bf   = (unsigned short*)alloc((size_t)N * D * 2);
    float*          aggf     = (float*)alloc((size_t)N * D * 4);
    float*          cntf     = (float*)alloc((size_t)N * 4);
    unsigned short* cat_bf   = (unsigned short*)alloc((size_t)N * 896 * 2);
    unsigned short* catpl_bf = (unsigned short*)alloc((size_t)N * 512 * 2);
    unsigned short* xs_bf    = (unsigned short*)alloc((size_t)K * N * D * 2);
    unsigned short* hl_bf    = (unsigned short*)alloc((size_t)N * D * 2);
    float*          cl_f     = (float*)alloc((size_t)N * D * 4);

    auto pack = [&](const float* s, int Kd, int Nc, int tr, unsigned int* dpk) {
        int tiles = (Kd >> 5) * (Nc >> 4);
        k_pack_b<<<(tiles + 3) / 4, 128, 0, stream>>>(s, Kd, Nc, tr, dpk);
    };
    // weight packing (every call: deterministic, no caching)
    for (int i = 0; i < NCONV; ++i) {
        pack(Wp    + (size_t)i * D * D, 128, 128, 0, Wp_pk    + (size_t)i * 8192);
        pack(Wrel  + (size_t)i * D * D, 128, 128, 0, Wrel_pk  + (size_t)i * 8192);
        pack(Wroot + (size_t)i * D * D, 128, 128, 0, Wroot_pk + (size_t)i * 8192);
    }
    pack(Wcomb, 896, 384, 0, Wcomb_pk);
    pack(Wpl,   512, 128, 0, Wpl_pk);
    pack(Wih,   128, 512, 1, Wih_pk);   // logical B = Wih^T
    pack(Whh,   128, 512, 1, Whh_pk);   // logical B = Whh^T

    // LSTM state init
    k_zero_u16<<<(unsigned)(((size_t)N * D + 255) / 256), 256, 0, stream>>>(hl_bf, (size_t)N * D);
    k_zero_f32<<<(unsigned)(((size_t)N * D + 255) / 256), 256, 0, stream>>>(cl_f, (size_t)N * D);

    dim3 blk(128);
    for (int k = 0; k < K; ++k) {
        const float* Hk  = H + (size_t)k * N * D;
        const int*   src = el + (size_t)k * 2 * E;
        const int*   dst = src + E;

        k_prep_Hk<<<(unsigned)(((size_t)N * D + 255) / 256), 256, 0, stream>>>(
            Hk, Hk_bf, cat_bf, catpl_bf, (size_t)N * D);

        for (int i = 0; i < NCONV; ++i) {
            // h = relu(Hk @ Wp[i] + bp[i])  -> bf16 [N,128]
            k_gemm_bf16<<<dim3(MT, 1), blk, 0, stream>>>(
                Hk_bf, 128, Wp_pk + (size_t)i * 8192,
                nullptr, nullptr, bp + (size_t)i * D, 1,
                hproj_bf, 128, 0, 128, 128);

            k_zero_f32<<<(unsigned)(((size_t)N * D + 255) / 256), 256, 0, stream>>>(aggf, (size_t)N * D);
            k_zero_f32<<<(N + 255) / 256, 256, 0, stream>>>(cntf, (size_t)N);

            const float* ew = spec + ((size_t)k * NCONV + i) * E;
            k_edge_scatter<<<(E + 7) / 8, 256, 0, stream>>>(src, dst, ew, hproj_bf, aggf, cntf, E);

            k_normalize<<<N, 128, 0, stream>>>(aggf, cntf, agg_bf);

            // cat[:, (i+1)*128 : (i+2)*128] = agg @ Wrel[i] + Hk @ Wroot[i]
            k_gemm_bf16<<<dim3(MT, 1), blk, 0, stream>>>(
                agg_bf, 128, Wrel_pk + (size_t)i * 8192,
                Hk_bf, Wroot_pk + (size_t)i * 8192, nullptr, 0,
                cat_bf, 896, (i + 1) * 128, 128, 128);
        }

        // catpl[:,128:512] = relu(cat @ W_comb + b_comb)   (N x 896 @ 896 x 384)
        k_gemm_bf16<<<dim3(MT, 3), blk, 0, stream>>>(
            cat_bf, 896, Wcomb_pk, nullptr, nullptr, bcomb, 1,
            catpl_bf, 512, 128, 384, 896);

        // xs[k] = catpl @ W_pl + b_pl   (N x 512 @ 512 x 128)
        k_gemm_bf16<<<dim3(MT, 1), blk, 0, stream>>>(
            catpl_bf, 512, Wpl_pk, nullptr, nullptr, bpl, 0,
            xs_bf + (size_t)k * N * D, 128, 0, 128, 512);
    }

    // LSTM over K=3 steps; last step writes relu(h) to d_out
    for (int t = 0; t < K; ++t) {
        k_lstm_step<<<MT, 128, 0, stream>>>(
            xs_bf + (size_t)t * N * D, Wih_pk, Whh_pk, bih, bhh,
            hl_bf, cl_f, out, (t == K - 1) ? 1 : 0);
    }
}
